// energy_point_pointnet2_19842748908345
// MI455X (gfx1250) — matrix operand order repaired and checked by reading, not compiled, not run on hardware
//
#include <hip/hip_runtime.h>

#define DIV_UP(a,b) (((a)+(b)-1)/(b))

typedef __attribute__((ext_vector_type(16))) _Float16 v16h;
typedef __attribute__((ext_vector_type(8)))  _Float16 v8h;
typedef __attribute__((ext_vector_type(8)))  float    v8f;

// ----------------------------------------------------------------------------
// Weight pack: Wp[n*Kp + k] = fp16( transB ? W[n*K+k] : W[k*N+n] ), zero-padded
// in k. Kp-contiguous per output column -> B fragments are 128-bit loads.
// ----------------------------------------------------------------------------
__global__ void pack_w(const float* __restrict__ W, _Float16* __restrict__ Wp,
                       int K, int N, int Kp, int transB)
{
  const long total = (long)N * Kp;
  for (long i = (long)blockIdx.x * blockDim.x + threadIdx.x; i < total;
       i += (long)gridDim.x * blockDim.x) {
    int  k = (int)(i % Kp);
    long n = i / Kp;
    float v = 0.0f;
    if (k < K) v = transB ? W[(size_t)n * K + k] : W[(size_t)k * N + n];
    Wp[i] = (_Float16)v;
  }
}

// B 16-bit fragment: lane<16 holds K 0..15, lane>=16 holds K 16..31 (wo = k0+half*16)
__device__ __forceinline__ v16h load_bfrag(const _Float16* __restrict__ Wc, int wo)
{
  const v8h b0 = *(const v8h*)(Wc + wo);
  const v8h b1 = *(const v8h*)(Wc + wo + 8);
  v16h b;
#pragma unroll
  for (int j = 0; j < 8; ++j) { b[j] = b0[j]; b[8 + j] = b1[j]; }
  return b;
}

__device__ __forceinline__ void store_tile(float* __restrict__ Y, int N, int rowsM,
                                           int tm, int half, int col,
                                           const float* __restrict__ bias, int relu, v8f c)
{
  const float bb = bias ? bias[col] : 0.0f;
#pragma unroll
  for (int r = 0; r < 8; ++r) {
    int row = tm * 16 + r + (half ? 8 : 0);   // C/D: VGPR r -> M=r (+8 hi lanes)
    float v = c[r] + bb;
    if (relu) v = fmaxf(v, 0.0f);
    if (row >= rowsM) v = 0.0f;               // keep pad rows zero for consumers
    Y[(size_t)row * N + col] = v;
  }
}

// ----------------------------------------------------------------------------
// WMMA GEMM: Y = act( X[rowsP,Kp] @ Wp (+bias) ), fp16 operands, fp32 accum.
// One wave computes a 16x64 strip: A fragment loaded once per K-step and
// reused by 4 back-to-back WMMAs into 4 accumulators (grid.y = N/64).
// X padded: leading dim Kp (mult of 32); pad rows of Y written as zeros.
// ----------------------------------------------------------------------------
__global__ void gemm_wmma(const float* __restrict__ X, const _Float16* __restrict__ Wp,
                          const float* __restrict__ bias, float* __restrict__ Y,
                          int rowsM, int Kp, int N, int relu)
{
  const int lane = threadIdx.x & 31;
  const int half = lane >> 4;
  const int l16  = lane & 15;
  const int tm = blockIdx.x, tn = blockIdx.y;
  const int rowA = tm * 16 + l16;   // A: lane holds row M = lane%16
  const int colB = tn * 64 + l16;   // base column of the 4 tiles
  const float*    Xr  = X  + (size_t)rowA * Kp;
  const _Float16* Wc0 = Wp + (size_t)(colB     ) * Kp;
  const _Float16* Wc1 = Wp + (size_t)(colB + 16) * Kp;
  const _Float16* Wc2 = Wp + (size_t)(colB + 32) * Kp;
  const _Float16* Wc3 = Wp + (size_t)(colB + 48) * Kp;
  v8f c0 = {}, c1 = {}, c2 = {}, c3 = {};
  for (int k0 = 0; k0 < Kp; k0 += 32) {
    // A 16-bit layout: lane<16 -> K 0..15 ; lane>=16 -> K 16..31
    const float4 a0 = *(const float4*)(Xr + k0 + half * 16);
    const float4 a1 = *(const float4*)(Xr + k0 + half * 16 + 4);
    const float4 a2 = *(const float4*)(Xr + k0 + half * 16 + 8);
    const float4 a3 = *(const float4*)(Xr + k0 + half * 16 + 12);
    v16h a;
    a[0]  = (_Float16)a0.x; a[1]  = (_Float16)a0.y; a[2]  = (_Float16)a0.z; a[3]  = (_Float16)a0.w;
    a[4]  = (_Float16)a1.x; a[5]  = (_Float16)a1.y; a[6]  = (_Float16)a1.z; a[7]  = (_Float16)a1.w;
    a[8]  = (_Float16)a2.x; a[9]  = (_Float16)a2.y; a[10] = (_Float16)a2.z; a[11] = (_Float16)a2.w;
    a[12] = (_Float16)a3.x; a[13] = (_Float16)a3.y; a[14] = (_Float16)a3.z; a[15] = (_Float16)a3.w;
    const int wo = k0 + half * 16;
    v16h b;
    b  = load_bfrag(Wc0, wo);
    c0 = __builtin_amdgcn_wmma_f32_16x16x32_f16(false, a, false, b, (short)0, c0, false, false);
    b  = load_bfrag(Wc1, wo);
    c1 = __builtin_amdgcn_wmma_f32_16x16x32_f16(false, a, false, b, (short)0, c1, false, false);
    b  = load_bfrag(Wc2, wo);
    c2 = __builtin_amdgcn_wmma_f32_16x16x32_f16(false, a, false, b, (short)0, c2, false, false);
    b  = load_bfrag(Wc3, wo);
    c3 = __builtin_amdgcn_wmma_f32_16x16x32_f16(false, a, false, b, (short)0, c3, false, false);
  }
  store_tile(Y, N, rowsM, tm, half, colB,      bias, relu, c0);
  store_tile(Y, N, rowsM, tm, half, colB + 16, bias, relu, c1);
  store_tile(Y, N, rowsM, tm, half, colB + 32, bias, relu, c2);
  store_tile(Y, N, rowsM, tm, half, colB + 48, bias, relu, c3);
}

// ----------------------------------------------------------------------------
// Farthest point sampling. One block per batch; distance array in LDS.
// Deterministic start at index 0; ties -> lowest index (matches jnp.argmax).
// ----------------------------------------------------------------------------
#define MAXN 4096
__global__ void fps_kernel(const float* __restrict__ pos, int* __restrict__ idx,
                           int N, int M)
{
  __shared__ float d[MAXN];
  __shared__ float rv[256];
  __shared__ int   ri[256];
  __shared__ int   s_next;
  const int b = blockIdx.x, t = threadIdx.x;
  const float* P = pos + (size_t)b * N * 3;
  const float x0 = P[0], y0 = P[1], z0 = P[2];
  for (int i = t; i < N; i += 256) {
    float dx = P[i*3] - x0, dy = P[i*3+1] - y0, dz = P[i*3+2] - z0;
    d[i] = dx*dx + dy*dy + dz*dz;
  }
  if (t == 0) idx[(size_t)b * M] = 0;
  __syncthreads();
  for (int s = 1; s < M; ++s) {
    float best = -1.0f; int bi = 0;
    for (int i = t; i < N; i += 256) {
      float v = d[i];
      if (v > best) { best = v; bi = i; }
    }
    rv[t] = best; ri[t] = bi;
    __syncthreads();
    for (int stp = 128; stp > 0; stp >>= 1) {
      if (t < stp) {
        if (rv[t+stp] > rv[t] || (rv[t+stp] == rv[t] && ri[t+stp] < ri[t])) {
          rv[t] = rv[t+stp]; ri[t] = ri[t+stp];
        }
      }
      __syncthreads();
    }
    if (t == 0) { s_next = ri[0]; idx[(size_t)b * M + s] = ri[0]; }
    __syncthreads();
    const int nx = s_next;
    const float xn = P[nx*3], yn = P[nx*3+1], zn = P[nx*3+2];
    for (int i = t; i < N; i += 256) {
      float dx = P[i*3] - xn, dy = P[i*3+1] - yn, dz = P[i*3+2] - zn;
      float nd = dx*dx + dy*dy + dz*dz;
      if (nd < d[i]) d[i] = nd;
    }
    __syncthreads();
  }
}

__global__ void gather_centers(const float* __restrict__ pos, const int* __restrict__ idx,
                               float* __restrict__ centers, int B, int N, int M)
{
  int i = blockIdx.x * blockDim.x + threadIdx.x;
  if (i >= B * M) return;
  int b = i / M;
  int n = idx[i];
  const float* P = pos + ((size_t)b * N + n) * 3;
  centers[(size_t)i*3 + 0] = P[0];
  centers[(size_t)i*3 + 1] = P[1];
  centers[(size_t)i*3 + 2] = P[2];
}

// ----------------------------------------------------------------------------
// Radius query, K nearest-first within r^2. One block per (center, batch).
// Outside-radius slots get mask=0 (their indices are unused by BN / max-agg).
// ----------------------------------------------------------------------------
__global__ void radius_kernel(const float* __restrict__ pos, const float* __restrict__ centers,
                              int* __restrict__ nbr, float* __restrict__ mask,
                              int N, int M, int K, float r2)
{
  __shared__ float d[MAXN];
  __shared__ float rv[256];
  __shared__ int   ri[256];
  const int m = blockIdx.x, b = blockIdx.y, t = threadIdx.x;
  const float* P  = pos + (size_t)b * N * 3;
  const float* Cc = centers + ((size_t)b * M + m) * 3;
  const float cx = Cc[0], cy = Cc[1], cz = Cc[2];
  for (int i = t; i < N; i += 256) {
    float dx = P[i*3] - cx, dy = P[i*3+1] - cy, dz = P[i*3+2] - cz;
    d[i] = dx*dx + dy*dy + dz*dz;
  }
  __syncthreads();
  const size_t base = ((size_t)b * M + m) * K;
  for (int s = 0; s < K; ++s) {
    float best = 3.4e38f; int bi = 0;
    for (int i = t; i < N; i += 256) {
      float v = d[i];
      if (v < best) { best = v; bi = i; }
    }
    rv[t] = best; ri[t] = bi;
    __syncthreads();
    for (int stp = 128; stp > 0; stp >>= 1) {
      if (t < stp) {
        if (rv[t+stp] < rv[t] || (rv[t+stp] == rv[t] && ri[t+stp] < ri[t])) {
          rv[t] = rv[t+stp]; ri[t] = ri[t+stp];
        }
      }
      __syncthreads();
    }
    if (t == 0) {
      int sel = ri[0];
      nbr[base + s]  = sel;
      mask[base + s] = (rv[0] <= r2) ? 1.0f : 0.0f;
      d[sel] = 3.4e38f;
    }
    __syncthreads();
  }
}

// feat[b,m,k,:] = concat( x[b,nbr,:CX], pos[b,nbr]-center[b,m] ), zero-padded
// to leading dim ldF (multiple of 32) for the padded WMMA GEMM.
__global__ void build_feat(const float* __restrict__ pos, const float* __restrict__ centers,
                           const float* __restrict__ x, const int* __restrict__ nbr,
                           float* __restrict__ feat, int B, int N, int M, int K,
                           int CX, int ldF)
{
  const long total = (long)B * M * K;
  for (long row = (long)blockIdx.x * blockDim.x + threadIdx.x; row < total;
       row += (long)gridDim.x * blockDim.x) {
    long bm = row / K;
    int  m  = (int)(bm % M);
    int  b  = (int)(bm / M);
    int n = nbr[row];
    const float* Pn = pos + ((size_t)b * N + n) * 3;
    const float* Cc = centers + ((size_t)b * M + m) * 3;
    float* F = feat + (size_t)row * ldF;
    if (CX) {
      const float* Xr = x + ((size_t)b * N + n) * CX;
      for (int c = 0; c < CX; ++c) F[c] = Xr[c];
    }
    F[CX+0] = Pn[0] - Cc[0];
    F[CX+1] = Pn[1] - Cc[1];
    F[CX+2] = Pn[2] - Cc[2];
    for (int c = CX + 3; c < ldF; ++c) F[c] = 0.0f;
  }
}

// Masked BatchNorm stats: one block per channel.
__global__ void bn_stats(const float* __restrict__ h, const float* __restrict__ mask,
                         float* __restrict__ meanv, float* __restrict__ istd,
                         long rows, int C)
{
  const int c = blockIdx.x, t = threadIdx.x;
  __shared__ float ssum[256], ssq[256], scnt[256];
  float s = 0.f, q = 0.f, n = 0.f;
  for (long i = t; i < rows; i += 256) {
    float mv = mask ? mask[i] : 1.0f;
    float v  = h[(size_t)i * C + c];
    s += v * mv; q += v * v * mv; n += mv;
  }
  ssum[t] = s; ssq[t] = q; scnt[t] = n;
  __syncthreads();
  for (int stp = 128; stp > 0; stp >>= 1) {
    if (t < stp) { ssum[t] += ssum[t+stp]; ssq[t] += ssq[t+stp]; scnt[t] += scnt[t+stp]; }
    __syncthreads();
  }
  if (t == 0) {
    float cnt = scnt[0];
    float mu  = ssum[0] / cnt;
    float var = ssq[0] / cnt - mu * mu;
    meanv[c] = mu;
    istd[c]  = rsqrtf(var + 1e-5f);
  }
}

__global__ void bn_apply(float* __restrict__ h, const float* __restrict__ meanv,
                         const float* __restrict__ istd, const float* __restrict__ g,
                         const float* __restrict__ beta, long rows, int C)
{
  const long total = rows * C;
  for (long i = (long)blockIdx.x * blockDim.x + threadIdx.x; i < total;
       i += (long)gridDim.x * blockDim.x) {
    int c = (int)(i % C);
    h[i] = g[c] * (h[i] - meanv[c]) * istd[c] + beta[c];
  }
}

// out[bm,c] = max_k ( mask[bm,k] ? h[bm,k,c] : -inf );  mask may be null.
__global__ void max_agg(const float* __restrict__ h, const float* __restrict__ mask,
                        float* __restrict__ out, int BM, int K, int C)
{
  const long total = (long)BM * C;
  for (long i = (long)blockIdx.x * blockDim.x + threadIdx.x; i < total;
       i += (long)gridDim.x * blockDim.x) {
    long bm = i / C;
    int  c  = (int)(i % C);
    float best = -3.4e38f;
    const float* hp = h + ((size_t)bm * K) * C + c;
    for (int k = 0; k < K; ++k) {
      if (!mask || mask[(size_t)bm * K + k] != 0.0f) {
        float v = hp[(size_t)k * C];
        if (v > best) best = v;
      }
    }
    out[i] = best;
  }
}

// ----------------------------------------------------------------------------
// Streaming online-softmax attention, head dim d == 32. One wave per
// (batch, head, query). Each lane owns a strided subset of keys with a local
// (max, sum, acc[32]); merged across lanes via LDS. qkv layout [B*N, 3C].
// ----------------------------------------------------------------------------
__global__ void attn_kernel(const float* __restrict__ qkv, float* __restrict__ o,
                            int B, int N, int C, int H)
{
  const int n = blockIdx.x, hh = blockIdx.y, b = blockIdx.z;
  const int lane = threadIdx.x;
  const float scale = rsqrtf(32.0f);
  const float* qrow = qkv + ((size_t)b * N + n) * 3 * C + hh * 32;
  float q[32];
#pragma unroll
  for (int j = 0; j < 32; ++j) q[j] = qrow[j] * scale;
  float mloc = -3.4e38f, sloc = 0.0f;
  float acc[32];
#pragma unroll
  for (int j = 0; j < 32; ++j) acc[j] = 0.0f;
  for (int m = lane; m < N; m += 32) {
    const float* krow = qkv + ((size_t)b * N + m) * 3 * C + C + hh * 32;
    float dot = 0.0f;
#pragma unroll
    for (int j = 0; j < 32; ++j) dot += q[j] * krow[j];
    float mnew = fmaxf(mloc, dot);
    float corr = __expf(mloc - mnew);
    float p    = __expf(dot - mnew);
    sloc = sloc * corr + p;
    const float* vrow = qkv + ((size_t)b * N + m) * 3 * C + 2 * C + hh * 32;
#pragma unroll
    for (int j = 0; j < 32; ++j) acc[j] = acc[j] * corr + p * vrow[j];
    mloc = mnew;
  }
  __shared__ float sm[32], ss[32], sa[32][33];
  sm[lane] = mloc; ss[lane] = sloc;
#pragma unroll
  for (int j = 0; j < 32; ++j) sa[lane][j] = acc[j];
  __syncthreads();
  float gm = -3.4e38f;
  for (int l = 0; l < 32; ++l) gm = fmaxf(gm, sm[l]);
  float den = 0.0f, num = 0.0f;
  for (int l = 0; l < 32; ++l) {
    float w = __expf(sm[l] - gm);
    den += w * ss[l];
    num += w * sa[l][lane];
  }
  o[((size_t)b * N + n) * C + hh * 32 + lane] = num / den;
}

// In-place residual + LayerNorm: x = LN(x + res). One 128-thread block per row.
__global__ void ln_residual(float* __restrict__ x, const float* __restrict__ res,
                            const float* __restrict__ g, const float* __restrict__ bta,
                            int C)
{
  const long row = blockIdx.x;
  const int t = threadIdx.x;
  __shared__ float s1[128], s2[128];
  float* xr = x + row * C;
  const float* rr = res + row * C;
  float s = 0.f, q = 0.f;
  for (int c = t; c < C; c += 128) {
    float v = xr[c] + rr[c];
    s += v; q += v * v;
  }
  s1[t] = s; s2[t] = q;
  __syncthreads();
  for (int stp = 64; stp > 0; stp >>= 1) {
    if (t < stp) { s1[t] += s1[t+stp]; s2[t] += s2[t+stp]; }
    __syncthreads();
  }
  float mu  = s1[0] / C;
  float var = s2[0] / C - mu * mu;
  float is  = rsqrtf(var + 1e-5f);
  for (int c = t; c < C; c += 128) {
    float v = xr[c] + rr[c];
    xr[c] = g[c] * (v - mu) * is + bta[c];
  }
}

// out[row, 0..CX) = x[row], out[row, CX..CX+3) = p[row], zero pad to ld.
__global__ void concat_xp(const float* __restrict__ x, const float* __restrict__ p,
                          float* __restrict__ out, int rows, int CX, int ld)
{
  const long total = (long)rows * ld;
  for (long i = (long)blockIdx.x * blockDim.x + threadIdx.x; i < total;
       i += (long)gridDim.x * blockDim.x) {
    long row = i / ld;
    int  c   = (int)(i % ld);
    float v = 0.0f;
    if (c < CX)          v = x[(size_t)row * CX + c];
    else if (c < CX + 3) v = p[(size_t)row * 3 + (c - CX)];
    out[i] = v;
  }
}

__global__ void zero_fill(float* __restrict__ p, long n)
{
  for (long i = (long)blockIdx.x * blockDim.x + threadIdx.x; i < n;
       i += (long)gridDim.x * blockDim.x) p[i] = 0.0f;
}

// Tiny final linear (N=10): one thread per output element.
__global__ void linear_small(const float* __restrict__ x, const float* __restrict__ W,
                             const float* __restrict__ bias, float* __restrict__ y,
                             int rowsM, int K, int N)
{
  int i = blockIdx.x * blockDim.x + threadIdx.x;
  if (i >= rowsM * N) return;
  int row = i / N, n = i % N;
  float s = bias[n];
  for (int k = 0; k < K; ++k) s += x[(size_t)row * K + k] * W[(size_t)k * N + n];
  y[i] = s;
}

// ----------------------------------------------------------------------------
// Host orchestration
// ----------------------------------------------------------------------------
static inline int igrid(long total, int threads, int cap)
{
  long b = (total + threads - 1) / threads;
  if (b > cap) b = cap;
  if (b < 1) b = 1;
  return (int)b;
}

static void run_gemm(const float* X, int Kp, const float* W, int K, int N, int transB,
                     const float* bias, float* Y, long rows, int relu,
                     _Float16* wpack, hipStream_t s)
{
  pack_w<<<igrid((long)N * Kp, 256, 8192), 256, 0, s>>>(W, wpack, K, N, Kp, transB);
  dim3 grid(DIV_UP((int)rows, 16), N / 64);   // all N here are multiples of 64
  gemm_wmma<<<grid, 32, 0, s>>>(X, wpack, bias, Y, (int)rows, Kp, N, relu);
}

static void run_mlp_layer(const float* in, int Kp, float* out,
                          const float* W, const float* bW, const float* g, const float* beta,
                          const float* mask, long rows, int K, int N,
                          _Float16* wpack, float* meanb, float* istdb, hipStream_t s)
{
  run_gemm(in, Kp, W, K, N, 0, bW, out, rows, 1, wpack, s);
  bn_stats<<<N, 256, 0, s>>>(out, mask, meanb, istdb, rows, N);
  bn_apply<<<igrid(rows * (long)N, 256, 32768), 256, 0, s>>>(out, meanb, istdb, g, beta,
                                                             rows, N);
}

static void run_tf(float* x, long rowsT, int B, int N, int C, int H,
                   const float* Wqkv, const float* bqkv, const float* Wo, const float* bo,
                   const float* ln1g, const float* ln1b,
                   const float* W1, const float* b1, const float* W2, const float* b2,
                   const float* ln2g, const float* ln2b,
                   _Float16* wpack, float* buf0, float* buf1, hipStream_t s)
{
  run_gemm(x, C, Wqkv, C, 3 * C, 1, bqkv, buf0, rowsT, 0, wpack, s);   // qkv = x@Wqkv^T
  dim3 ga(N, H, B);
  attn_kernel<<<ga, 32, 0, s>>>(buf0, buf1, B, N, C, H);
  run_gemm(buf1, C, Wo, C, C, 1, bo, buf0, rowsT, 0, wpack, s);        // proj = o@Wo^T
  ln_residual<<<(int)rowsT, 128, 0, s>>>(x, buf0, ln1g, ln1b, C);
  run_gemm(x, C, W1, C, 2048, 0, b1, buf0, rowsT, 1, wpack, s);        // FFN up + ReLU
  run_gemm(buf0, 2048, W2, 2048, C, 0, b2, buf1, rowsT, 0, wpack, s);  // FFN down
  ln_residual<<<(int)rowsT, 128, 0, s>>>(x, buf1, ln2g, ln2b, C);
}

extern "C" void kernel_launch(void* const* d_in, const int* in_sizes, int n_in,
                              void* d_out, int out_size, void* d_ws, size_t ws_size,
                              hipStream_t stream)
{
  (void)in_sizes; (void)n_in; (void)out_size; (void)ws_size;
  // Input pytree flattened in JAX sorted-key order:
  // 0:data  1,2:lin1.{W,b}  3,4:lin2  5,6:lin3
  // sa1 L0..L2 each {W,b,beta,g}: 7..18 ; sa2: 19..30 ; sa3: 31..42
  // tf1 {W1,W2,Wo,Wqkv,b1,b2,bo,bqkv,ln1b,ln1g,ln2b,ln2g}: 43..54 ; tf2: 55..66
  const float* data = (const float*)d_in[0];
  #define F(i) ((const float*)d_in[(i)])

  const int B = 4, N0 = 4096, M1 = 2048, M2 = 512, K = 64;
  const long rows1 = (long)B * M1 * K;   // 524288 (mult of 16)
  const long rows2 = (long)B * M2 * K;   // 131072
  const long rows3 = (long)B * M2;       // 2048

  char* ws = (char*)d_ws;
  float* BUF0 = (float*)ws;                                  // 524288*128 f
  float* BUF1 = (float*)(ws + (size_t)rows1 * 128 * 4);      // 524288*128 f
  char*  sm   = ws + (size_t)2 * rows1 * 128 * 4;
  int*      idx1  = (int*)sm;             sm += (size_t)B * M1 * 4;
  float*    cent1 = (float*)sm;           sm += (size_t)B * M1 * 3 * 4;
  int*      nbr1  = (int*)sm;             sm += (size_t)B * M1 * K * 4;
  float*    mask1 = (float*)sm;           sm += (size_t)B * M1 * K * 4;
  float*    x1    = (float*)sm;           sm += (size_t)B * M1 * 128 * 4;
  int*      idx2  = (int*)sm;             sm += (size_t)B * M2 * 4;
  float*    cent2 = (float*)sm;           sm += (size_t)B * M2 * 3 * 4;
  int*      nbr2  = (int*)sm;             sm += (size_t)B * M2 * K * 4;
  float*    mask2 = (float*)sm;           sm += (size_t)B * M2 * K * 4;
  float*    x2    = (float*)sm;           sm += (size_t)B * M2 * 256 * 4;
  float*    gpool = (float*)sm;           sm += (size_t)16 * 1024 * 4;  // 16 pad rows
  float*    h1    = (float*)sm;           sm += (size_t)16 * 512 * 4;
  float*    h2    = (float*)sm;           sm += (size_t)16 * 256 * 4;
  float*    meanb = (float*)sm;           sm += 2048 * 4;
  float*    istdb = (float*)sm;           sm += 2048 * 4;
  _Float16* wpack = (_Float16*)sm;        sm += (size_t)1048576 * 2;    // max N*Kp

  // ---------------- SA1: FPS -> radius -> PointConv MLP -> max ----------------
  fps_kernel<<<B, 256, 0, stream>>>(data, idx1, N0, M1);
  gather_centers<<<DIV_UP(B * M1, 256), 256, 0, stream>>>(data, idx1, cent1, B, N0, M1);
  radius_kernel<<<dim3(M1, B), 256, 0, stream>>>(data, cent1, nbr1, mask1, N0, M1, K, 0.04f);
  build_feat<<<igrid(rows1, 256, 32768), 256, 0, stream>>>(data, cent1, nullptr, nbr1,
                                                           BUF0, B, N0, M1, K, 0, 32);
  run_mlp_layer(BUF0,  32, BUF1, F(7),  F(8),  F(10), F(9),  mask1, rows1,   3,  64, wpack, meanb, istdb, stream);
  run_mlp_layer(BUF1,  64, BUF0, F(11), F(12), F(14), F(13), mask1, rows1,  64,  64, wpack, meanb, istdb, stream);
  run_mlp_layer(BUF0,  64, BUF1, F(15), F(16), F(18), F(17), mask1, rows1,  64, 128, wpack, meanb, istdb, stream);
  max_agg<<<igrid((long)B * M1 * 128, 256, 32768), 256, 0, stream>>>(BUF1, mask1, x1,
                                                                    B * M1, K, 128);
  // ---------------- TF1 (C=128, H=4) ----------------
  run_tf(x1, (long)B * M1, B, M1, 128, 4,
         F(46), F(50), F(45), F(49), F(52), F(51),
         F(43), F(47), F(44), F(48), F(54), F(53), wpack, BUF0, BUF1, stream);

  // ---------------- SA2 ----------------
  fps_kernel<<<B, 256, 0, stream>>>(cent1, idx2, M1, M2);
  gather_centers<<<DIV_UP(B * M2, 256), 256, 0, stream>>>(cent1, idx2, cent2, B, M1, M2);
  radius_kernel<<<dim3(M2, B), 256, 0, stream>>>(cent1, cent2, nbr2, mask2, M1, M2, K, 0.16f);
  build_feat<<<igrid(rows2, 256, 32768), 256, 0, stream>>>(cent1, cent2, x1, nbr2,
                                                           BUF0, B, M1, M2, K, 128, 160);
  run_mlp_layer(BUF0, 160, BUF1, F(19), F(20), F(22), F(21), mask2, rows2, 131, 128, wpack, meanb, istdb, stream);
  run_mlp_layer(BUF1, 128, BUF0, F(23), F(24), F(26), F(25), mask2, rows2, 128, 128, wpack, meanb, istdb, stream);
  run_mlp_layer(BUF0, 128, BUF1, F(27), F(28), F(30), F(29), mask2, rows2, 128, 256, wpack, meanb, istdb, stream);
  max_agg<<<igrid((long)B * M2 * 256, 256, 32768), 256, 0, stream>>>(BUF1, mask2, x2,
                                                                    B * M2, K, 256);
  // ---------------- TF2 (C=256, H=8) ----------------
  run_tf(x2, (long)B * M2, B, M2, 256, 8,
         F(58), F(62), F(57), F(61), F(64), F(63),
         F(55), F(59), F(56), F(60), F(66), F(65), wpack, BUF0, BUF1, stream);

  // ---------------- SA3 (global MLP, no mask) + global max pool ----------------
  concat_xp<<<igrid(rows3 * 288, 256, 32768), 256, 0, stream>>>(x2, cent2, BUF0,
                                                                (int)rows3, 256, 288);
  run_mlp_layer(BUF0, 288, BUF1, F(31), F(32), F(34), F(33), nullptr, rows3, 259,  256, wpack, meanb, istdb, stream);
  run_mlp_layer(BUF1, 256, BUF0, F(35), F(36), F(38), F(37), nullptr, rows3, 256,  512, wpack, meanb, istdb, stream);
  run_mlp_layer(BUF0, 512, BUF1, F(39), F(40), F(42), F(41), nullptr, rows3, 512, 1024, wpack, meanb, istdb, stream);
  zero_fill<<<igrid(12 * 1024, 256, 64), 256, 0, stream>>>(gpool + 4 * 1024, 12 * 1024);
  max_agg<<<igrid((long)B * 1024, 256, 32768), 256, 0, stream>>>(BUF1, nullptr, gpool,
                                                                B, M2, 1024);
  // ---------------- Head (rows=4, padded to one 16-row tile) ----------------
  run_gemm(gpool, 1024, F(1), 1024, 512, 0, F(2), h1, 4, 1, wpack, stream);
  run_gemm(h1,     512, F(3),  512, 256, 0, F(4), h2, 4, 1, wpack, stream);
  linear_small<<<1, 64, 0, stream>>>(h2, F(5), F(6), (float*)d_out, B, 256, 10);
  #undef F
}
